// Model_52931176956034
// MI455X (gfx1250) — matrix operand order repaired and checked by reading, not compiled, not run on hardware
//
#include <hip/hip_runtime.h>
#include <math.h>
#include <stdint.h>

#define DEV __device__ __forceinline__

typedef __bf16 bf16_t;
typedef __bf16 v16bf __attribute__((ext_vector_type(16)));
typedef __bf16 v8bf  __attribute__((ext_vector_type(8)));
typedef float  v8f   __attribute__((ext_vector_type(8)));

// Problem constants
constexpr int Bc = 64, Sc = 100, Dc = 768, Hc = 8, HDc = 96, FFc = 3072;
constexpr int BSc = Bc * Sc;   // 6400 rows, divisible by 128
constexpr int SP  = 112;       // S padded to multiple of 16 (M/N tiles)
constexpr int SK  = 128;       // S padded to multiple of 32 (K dim of attn@V)
constexpr float INV_SQRT_HD = 0.10206207261596575f; // 1/sqrt(96)

// ---------------------------------------------------------------------------
// WMMA fragment helpers (layouts per CDNA5 ISA 7.12.2, bf16 16x16x32)
// A (16x32, row-major source, ld in elements):
//   lane = half*16 + lr ; element i<8  -> Atile[lr][half*8 + i]
//                         element i>=8 -> Atile[lr][16 + half*8 + (i-8)]
// B (32x16) sourced from Bt = B^T stored [N,K] row-major:
//   lane = half*16 + n  ; element i<8 -> Bt[n][half*8 + i], i>=8 -> Bt[n][16 + half*8 + (i-8)]
// C/D (16x16 f32): lane = half*16 + n ; vgpr v -> row (half*8 + v), col n
// ---------------------------------------------------------------------------
DEV v8f vzero8() { v8f z; for (int i = 0; i < 8; ++i) z[i] = 0.f; return z; }

DEV v16bf frag_a(const bf16_t* p, int ld, int half, int lr) {
  const bf16_t* q = p + (size_t)lr * ld + half * 8;
  v8bf lo = *(const v8bf*)q;
  v8bf hi = *(const v8bf*)(q + 16);
  return __builtin_shufflevector(lo, hi, 0,1,2,3,4,5,6,7,8,9,10,11,12,13,14,15);
}
DEV v16bf frag_b(const bf16_t* p, int ld, int half, int n) {
  const bf16_t* q = p + (size_t)n * ld + half * 8;
  v8bf lo = *(const v8bf*)q;
  v8bf hi = *(const v8bf*)(q + 16);
  return __builtin_shufflevector(lo, hi, 0,1,2,3,4,5,6,7,8,9,10,11,12,13,14,15);
}
DEV v8f wmma_bf(v16bf a, v16bf b, v8f c) {
  return __builtin_amdgcn_wmma_f32_16x16x32_bf16(false, a, false, b, (short)0, c, false, false);
}
DEV float gelu_exact(float x) { return 0.5f * x * (1.f + erff(x * 0.70710678118654752f)); }

// Async global->LDS copy, 16B per lane, tracked by ASYNCcnt (CDNA5 §10/§15.18)
DEV void async_cp_b128(unsigned lds_addr, const void* gaddr) {
  asm volatile("global_load_async_to_lds_b128 %0, %1, off"
               :: "v"(lds_addr), "v"(gaddr) : "memory");
}
DEV void wait_async0() { asm volatile("s_wait_asynccnt 0x0" ::: "memory"); }

// ---------------------------------------------------------------------------
// Dense bf16 WMMA GEMM with double-buffered async-LDS staging.
// C[M,N] = A[M,K] @ Bt[N,K]^T + bias ; WG tile 128x128 (8 waves as 4x2),
// wave tile 32x64, K-step 32. M,N multiples of 128; K multiple of 32.
// LDS: 2 stages x (A 128x40 + B 128x40) bf16 = 40 KB (stride 40 elem = 20
// banks -> conflict-free 16-row fragment gathers).
// EPI 0: fp32 out (+bias) ; EPI 1: gelu(x+bias) -> bf16 out
// ---------------------------------------------------------------------------
constexpr int LDP = 40;

template <int EPI>
__global__ __launch_bounds__(256) void k_gemm_lds(
    const bf16_t* __restrict__ A, int lda,
    const bf16_t* __restrict__ Bt, int ldb,
    const float* __restrict__ bias,
    void* __restrict__ Cout, int ldc, int K) {
  __shared__ __align__(16) bf16_t shA[2][128 * LDP];
  __shared__ __align__(16) bf16_t shB[2][128 * LDP];
  const int tid  = threadIdx.x;
  const int lane = tid & 31;
  const int w    = tid >> 5;
  const int half = lane >> 4;
  const int lr   = lane & 15;
  const int rowg = blockIdx.y * 128;
  const int colg = blockIdx.x * 128;
  const int rw   = (w >> 1) * 32;   // wave row offset in tile
  const int cw   = (w & 1) * 64;    // wave col offset in tile

  // staging assignment: thread copies 2 chunks of A and 2 of B (16B each)
  const int r0s = tid >> 2,          s0 = tid & 3;
  const int r1s = (tid + 256) >> 2,  s1 = (tid + 256) & 3;

  auto stage = [&](int buf, int k0) {
    async_cp_b128((unsigned)(uintptr_t)&shA[buf][r0s * LDP + s0 * 8],
                  A + (size_t)(rowg + r0s) * lda + k0 + s0 * 8);
    async_cp_b128((unsigned)(uintptr_t)&shA[buf][r1s * LDP + s1 * 8],
                  A + (size_t)(rowg + r1s) * lda + k0 + s1 * 8);
    async_cp_b128((unsigned)(uintptr_t)&shB[buf][r0s * LDP + s0 * 8],
                  Bt + (size_t)(colg + r0s) * ldb + k0 + s0 * 8);
    async_cp_b128((unsigned)(uintptr_t)&shB[buf][r1s * LDP + s1 * 8],
                  Bt + (size_t)(colg + r1s) * ldb + k0 + s1 * 8);
  };

  v8f acc[2][4];
  for (int i = 0; i < 2; ++i)
    for (int j = 0; j < 4; ++j) acc[i][j] = vzero8();

  stage(0, 0);
  const int nk = K / 32;
  for (int ks = 0; ks < nk; ++ks) {
    wait_async0();        // this wave's stage-ks copies have landed
    __syncthreads();      // all waves landed; previous compute buffer free
    if (ks + 1 < nk) stage((ks + 1) & 1, (ks + 1) * 32);  // overlap next DMA
    const bf16_t* sA = shA[ks & 1];
    const bf16_t* sB = shB[ks & 1];
    v16bf af[2], bfr[4];
    for (int i = 0; i < 2; ++i)
      af[i] = frag_a(sA + (size_t)(rw + 16 * i) * LDP, LDP, half, lr);
    for (int j = 0; j < 4; ++j)
      bfr[j] = frag_b(sB + (size_t)(cw + 16 * j) * LDP, LDP, half, lr);
    for (int i = 0; i < 2; ++i)
      for (int j = 0; j < 4; ++j) acc[i][j] = wmma_bf(af[i], bfr[j], acc[i][j]);
  }

  for (int i = 0; i < 2; ++i) {
    for (int j = 0; j < 4; ++j) {
      const int r0 = rowg + rw + 16 * i + half * 8;
      const int c  = colg + cw + 16 * j + lr;
      const float bv = bias ? bias[c] : 0.f;
      for (int v = 0; v < 8; ++v) {
        float val = acc[i][j][v] + bv;
        if (EPI == 0) {
          ((float*)Cout)[(size_t)(r0 + v) * ldc + c] = val;
        } else {
          ((bf16_t*)Cout)[(size_t)(r0 + v) * ldc + c] = (bf16_t)gelu_exact(val);
        }
      }
    }
  }
}

// ---------------------------------------------------------------------------
// Self-GEMM per (b,h): Out[b,h,i,j] = scale * E[b,i,h]·E[b,j,h] (+ addend)
// Stages the 100x96 E-slab into LDS once (zero-padded to 112 rows, stride 104
// -> 16B-aligned rows, conflict-free), then all fragments read unchecked.
// ---------------------------------------------------------------------------
__global__ __launch_bounds__(256) void k_scores(
    const bf16_t* __restrict__ E, const float* __restrict__ addend,
    float* __restrict__ Out, float scale) {
  __shared__ __align__(16) bf16_t sE[SP * 104];
  const int bh = blockIdx.x;
  const int b = bh >> 3, h = bh & 7;
  const bf16_t* base = E + (size_t)b * Sc * Dc + h * HDc;
  float* out = Out + (size_t)bh * SP * SP;
  const float* add = addend ? addend + (size_t)bh * SP * SP : nullptr;
  const int tid = threadIdx.x;
  const int lane = tid & 31, w = tid >> 5;
  const int half = lane >> 4, lr = lane & 15;

  for (int t = tid; t < SP * 12; t += 256) {   // 112 rows x 12 chunks of 16B
    const int row = t / 12, seg = t % 12;
    v8bf v;
    if (row < Sc) v = *(const v8bf*)(base + (size_t)row * Dc + seg * 8);
    else          for (int i = 0; i < 8; ++i) v[i] = (bf16_t)0.f;
    *(v8bf*)&sE[row * 104 + seg * 8] = v;
  }
  __syncthreads();

  for (int t = w; t < 49; t += 8) {  // 7x7 tiles of 16x16
    const int ti = t / 7, tj = t % 7;
    v8f acc = vzero8();
    for (int k0 = 0; k0 < HDc; k0 += 32) {
      v16bf a  = frag_a(sE + (size_t)(ti * 16) * 104 + k0, 104, half, lr);
      v16bf bb = frag_b(sE + (size_t)(tj * 16) * 104 + k0, 104, half, lr);
      acc = wmma_bf(a, bb, acc);
    }
    const int r0 = ti * 16 + half * 8;
    const int c  = tj * 16 + lr;
    for (int v = 0; v < 8; ++v) {
      const size_t idx = (size_t)(r0 + v) * SP + c;
      out[idx] = acc[v] * scale + (add ? add[idx] : 0.f);
    }
  }
}

// Softmax over valid 100 cols; writes bf16 attn padded to SK=128 with zeros.
__global__ __launch_bounds__(128) void k_softmax(
    const float* __restrict__ LOG, bf16_t* __restrict__ ATT) {
  const int w = threadIdx.x >> 5, lane = threadIdx.x & 31;
  const int R = blockIdx.x * 4 + w;          // R < B*H*SP
  const int i = R % SP, bh = R / SP;
  bf16_t* orow = ATT + (size_t)bh * SP * SK + (size_t)i * SK;
  if (i >= Sc) {
    for (int t = 0; t < 4; ++t) orow[lane + 32 * t] = (bf16_t)0.f;
    return;
  }
  const float* lrow = LOG + (size_t)bh * SP * SP + (size_t)i * SP;
  float v[4], mx = -1e30f;
  for (int t = 0; t < 4; ++t) {
    const int j = lane + 32 * t;
    v[t] = (j < Sc) ? lrow[j] : -1e30f;
    mx = fmaxf(mx, v[t]);
  }
  for (int off = 16; off > 0; off >>= 1) mx = fmaxf(mx, __shfl_xor(mx, off, 32));
  float e[4], sum = 0.f;
  for (int t = 0; t < 4; ++t) {
    e[t] = (lane + 32 * t < Sc) ? __expf(v[t] - mx) : 0.f;
    sum += e[t];
  }
  for (int off = 16; off > 0; off >>= 1) sum += __shfl_xor(sum, off, 32);
  const float inv = 1.f / sum;
  for (int t = 0; t < 4; ++t) {
    const int j = lane + 32 * t;
    orow[j] = (bf16_t)((j < Sc) ? e[t] * inv : 0.f);
  }
}

// attn @ V per (b,h): A = attn [SP,SK] bf16, Bt = qt [HD,SK] bf16 -> out fp32 [B,S,D]
__global__ __launch_bounds__(256) void k_av(
    const bf16_t* __restrict__ ATT, const bf16_t* __restrict__ QT,
    float* __restrict__ Out) {
  const int bh = blockIdx.x;
  const int b = bh >> 3, h = bh & 7;
  const bf16_t* A  = ATT + (size_t)bh * SP * SK;
  const bf16_t* Bt = QT + (size_t)bh * HDc * SK;
  const int lane = threadIdx.x & 31, w = threadIdx.x >> 5;
  const int half = lane >> 4, lr = lane & 15;
  for (int t = w; t < 42; t += 8) {  // 7 M-tiles x 6 N-tiles
    const int ti = t / 6, tj = t % 6;
    v8f acc = vzero8();
    for (int k0 = 0; k0 < SK; k0 += 32) {
      v16bf a  = frag_a(A + (size_t)(ti * 16) * SK + k0, SK, half, lr);
      v16bf bb = frag_b(Bt + (size_t)(tj * 16) * SK + k0, SK, half, lr);
      acc = wmma_bf(a, bb, acc);
    }
    const int r0 = ti * 16 + half * 8;
    const int c  = tj * 16 + lr;
    for (int v = 0; v < 8; ++v) {
      const int r = r0 + v;
      if (r < Sc)
        Out[((size_t)b * Sc + r) * Dc + h * HDc + c] = acc[v];
    }
  }
}

// LayerNorm over D=768 (+optional residual); optional fp32 and bf16 outputs.
__global__ __launch_bounds__(256) void k_ln(
    const float* __restrict__ A, const float* __restrict__ R,
    const float* __restrict__ g, const float* __restrict__ be,
    float* __restrict__ Of, bf16_t* __restrict__ Ob) {
  __shared__ float red[256];
  const int row = blockIdx.x, tid = threadIdx.x;
  const float* a = A + (size_t)row * Dc;
  const float* r = R ? R + (size_t)row * Dc : nullptr;
  float x[3], s = 0.f;
  for (int t = 0; t < 3; ++t) {
    const int d = tid + 256 * t;
    float v = a[d] + (r ? r[d] : 0.f);
    x[t] = v; s += v;
  }
  red[tid] = s; __syncthreads();
  for (int o = 128; o > 0; o >>= 1) { if (tid < o) red[tid] += red[tid + o]; __syncthreads(); }
  const float mean = red[0] * (1.f / 768.f);
  __syncthreads();
  float vs = 0.f;
  for (int t = 0; t < 3; ++t) { const float d2 = x[t] - mean; vs += d2 * d2; }
  red[tid] = vs; __syncthreads();
  for (int o = 128; o > 0; o >>= 1) { if (tid < o) red[tid] += red[tid + o]; __syncthreads(); }
  const float inv = rsqrtf(red[0] * (1.f / 768.f) + 1e-5f);
  for (int t = 0; t < 3; ++t) {
    const int d = tid + 256 * t;
    const float y = (x[t] - mean) * inv * g[d] + be[d];
    if (Of) Of[(size_t)row * Dc + d] = y;
    if (Ob) Ob[(size_t)row * Dc + d] = (bf16_t)y;
  }
}

// x = template + sinusoidal PE ; also emit bf16 copy
__global__ void k_pos(const float* __restrict__ T, float* __restrict__ X,
                      bf16_t* __restrict__ Xb) {
  const size_t idx = (size_t)blockIdx.x * 256 + threadIdx.x;
  if (idx >= (size_t)BSc * Dc) return;
  const int d = (int)(idx % Dc);
  const int s = (int)((idx / Dc) % Sc);
  const int i2 = d >> 1;
  const float ang = (float)s * __expf((float)(2 * i2) * (-9.210340371976184f / 768.f));
  const float pe = (d & 1) ? __cosf(ang) : __sinf(ang);
  const float v = T[idx] + pe;
  X[idx] = v;
  Xb[idx] = (bf16_t)v;
}

// pooled = mean over P=8 params -> bf16
__global__ void k_pool(const float* __restrict__ P, bf16_t* __restrict__ O) {
  const size_t idx = (size_t)blockIdx.x * 256 + threadIdx.x;
  if (idx >= (size_t)BSc * Dc) return;
  const size_t row = idx / Dc, d = idx % Dc;
  float s = 0.f;
  for (int p = 0; p < 8; ++p) s += P[(row * 8 + p) * Dc + d];
  O[idx] = (bf16_t)(s * 0.125f);
}

// transpose-convert fp32 W[K,N] -> bf16 Wt[N,K]
__global__ void k_wt(const float* __restrict__ W, bf16_t* __restrict__ Wt, int K, int N) {
  const size_t idx = (size_t)blockIdx.x * 256 + threadIdx.x;
  if (idx >= (size_t)K * N) return;
  const size_t k = idx / N, n = idx % N;
  Wt[n * (size_t)K + k] = (bf16_t)W[idx];
}

// qt[b,h,n,s] = xbf[b,s,h*96+n], zero-padded to s<128
__global__ void k_qt(const bf16_t* __restrict__ Xb, bf16_t* __restrict__ QT) {
  const size_t idx = (size_t)blockIdx.x * 256 + threadIdx.x;
  if (idx >= (size_t)Bc * Dc * SK) return;
  const int s = (int)(idx % SK);
  const size_t rem = idx / SK;
  const int d = (int)(rem % Dc);
  const int b = (int)(rem / Dc);
  const int h = d / HDc, n = d % HDc;
  bf16_t v = (s < Sc) ? Xb[((size_t)b * Sc + s) * Dc + d] : (bf16_t)0.f;
  QT[(((size_t)(b * Hc + h)) * HDc + n) * SK + s] = v;
}

// final classifier: out[b, 0..1] = x[b,:] . Wfc[:,0..1] + bfc
__global__ __launch_bounds__(256) void k_fc(
    const float* __restrict__ X, const float* __restrict__ Wfc,
    const float* __restrict__ bfc, float* __restrict__ out) {
  __shared__ float r0[256], r1[256];
  const int b = blockIdx.x, tid = threadIdx.x;
  const float* x = X + (size_t)b * (Sc * Dc);
  float p0 = 0.f, p1 = 0.f;
  for (int i = tid; i < Sc * Dc; i += 256) {
    const float xv = x[i];
    p0 += xv * Wfc[2 * i];
    p1 += xv * Wfc[2 * i + 1];
  }
  r0[tid] = p0; r1[tid] = p1; __syncthreads();
  for (int o = 128; o > 0; o >>= 1) {
    if (tid < o) { r0[tid] += r0[tid + o]; r1[tid] += r1[tid + o]; }
    __syncthreads();
  }
  if (tid == 0) {
    out[2 * b]     = r0[0] + bfc[0];
    out[2 * b + 1] = r1[0] + bfc[1];
  }
}

// ---------------------------------------------------------------------------
extern "C" void kernel_launch(void* const* d_in, const int* in_sizes, int n_in,
                              void* d_out, int out_size, void* d_ws, size_t ws_size,
                              hipStream_t stream) {
  const float* tseq = (const float*)d_in[0];
  const float* pseq = (const float*)d_in[1];
  const float* Wc   = (const float*)d_in[2];
  const float* bc   = (const float*)d_in[3];
  const float* gp   = (const float*)d_in[4];
  const float* bp   = (const float*)d_in[5];
  const float* W1   = (const float*)d_in[6];
  const float* b1   = (const float*)d_in[7];
  const float* W2   = (const float*)d_in[8];
  const float* b2   = (const float*)d_in[9];
  const float* g1   = (const float*)d_in[10];
  const float* bn1  = (const float*)d_in[11];
  const float* g2   = (const float*)d_in[12];
  const float* bn2  = (const float*)d_in[13];
  const float* Wfc  = (const float*)d_in[14];
  const float* bfc  = (const float*)d_in[15];

  char* ws = (char*)d_ws;
  size_t off = 0;
  auto carve = [&](size_t bytes) { char* p = ws + off; off += (bytes + 255) & ~(size_t)255; return p; };
  float*  X0   = (float*)carve((size_t)BSc * Dc * 4);         // x (ping)
  float*  X1   = (float*)carve((size_t)BSc * Dc * 4);         // src after LN1 (pong)
  bf16_t* XBF  = (bf16_t*)carve((size_t)BSc * Dc * 2);        // bf16 activations
  bf16_t* QT   = (bf16_t*)carve((size_t)Bc * Hc * HDc * SK * 2);
  float*  PB   = (float*)carve((size_t)Bc * Hc * SP * SP * 4); // layer-invariant bias
  float*  LOGI = (float*)carve((size_t)Bc * Hc * SP * SP * 4);
  bf16_t* ATT  = (bf16_t*)carve((size_t)Bc * Hc * SP * SK * 2);
  float*  AOUT = (float*)carve((size_t)BSc * Dc * 4);          // attn out / ffn out
  bf16_t* HFFN = (bf16_t*)carve((size_t)BSc * FFc * 2);
  bf16_t* WCT  = (bf16_t*)carve((size_t)Dc * Dc * 2);
  bf16_t* W1T  = (bf16_t*)carve((size_t)FFc * Dc * 2);
  bf16_t* W2T  = (bf16_t*)carve((size_t)Dc * FFc * 2);
  // pre-loop aliases (consumed before the layer loop writes them)
  bf16_t* POOL  = HFFN;   // pooled bf16 [6400,768]
  float*  PPROJ = AOUT;   // pooled@Wc fp32
  bf16_t* PENC  = XBF;    // layernormed param encoding bf16

  const int nElem = BSc * Dc;                      // 4,915,200
  const int gElem = (nElem + 255) / 256;

  // --- one-time weight prep ---
  k_wt<<<(Dc * Dc + 255) / 256, 256, 0, stream>>>(Wc, WCT, Dc, Dc);
  k_wt<<<(Dc * FFc + 255) / 256, 256, 0, stream>>>(W1, W1T, Dc, FFc);
  k_wt<<<(FFc * Dc + 255) / 256, 256, 0, stream>>>(W2, W2T, FFc, Dc);

  // --- layer-invariant param bias PB (computed once, reused 4x) ---
  k_pool<<<gElem, 256, 0, stream>>>(pseq, POOL);
  k_gemm_lds<0><<<dim3(Dc / 128, BSc / 128), 256, 0, stream>>>(POOL, Dc, WCT, Dc, bc, PPROJ, Dc, Dc);
  k_ln<<<BSc, 256, 0, stream>>>(PPROJ, nullptr, gp, bp, nullptr, PENC);
  k_scores<<<Bc * Hc, 256, 0, stream>>>(PENC, nullptr, PB, INV_SQRT_HD);

  // --- x = template + positional encoding ---
  k_pos<<<gElem, 256, 0, stream>>>(tseq, X0, XBF);

  for (int l = 0; l < 4; ++l) {
    k_qt<<<(Bc * Dc * SK + 255) / 256, 256, 0, stream>>>(XBF, QT);
    k_scores<<<Bc * Hc, 256, 0, stream>>>(XBF, PB, LOGI, INV_SQRT_HD);
    k_softmax<<<Bc * Hc * SP / 4, 128, 0, stream>>>(LOGI, ATT);
    k_av<<<Bc * Hc, 256, 0, stream>>>(ATT, QT, AOUT);
    k_ln<<<BSc, 256, 0, stream>>>(X0, AOUT, g1, bn1, X1, XBF);
    k_gemm_lds<1><<<dim3(FFc / 128, BSc / 128), 256, 0, stream>>>(XBF, Dc, W1T, Dc, b1, HFFN, FFc, Dc);
    k_gemm_lds<0><<<dim3(Dc / 128, BSc / 128), 256, 0, stream>>>(HFFN, FFc, W2T, FFc, b2, AOUT, Dc, FFc);
    k_ln<<<BSc, 256, 0, stream>>>(X1, AOUT, g2, bn2, X0, XBF);
  }

  k_fc<<<Bc, 256, 0, stream>>>(X0, Wfc, bfc, (float*)d_out);
}
